// DeepEmbed_69123203662023
// MI455X (gfx1250) — matrix operand order repaired and checked by reading, not compiled, not run on hardware
//
#include <hip/hip_runtime.h>
#include <hip/hip_bf16.h>

typedef __attribute__((ext_vector_type(16))) __bf16 v16bf;
typedef __attribute__((ext_vector_type(8)))  float  v8f;

// Problem constants (reference: VOCAB=50257, D_MODEL=2048, DEEP_DIM=64, B=4, S=4096)
#define NTOK   16384   // B*S
#define DMODEL 2048
#define KDIM   64

// Each block: 8 waves (256 threads), one 16-wide column tile of D_MODEL,
// 8 row tiles of 16 tokens each. W tile staged in LDS in WMMA-B layout.
__global__ __launch_bounds__(256) void deep_embed_wmma_kernel(
    const int*   __restrict__ token_ids,
    const float* __restrict__ x,
    const float* __restrict__ deep_table,
    const float* __restrict__ proj_w,
    float*       __restrict__ out)
{
    // B operand for v_wmma_f32_16x16x32_bf16, pre-swizzled:
    // index = kb*512 + lane*16 + h, where element is
    //   K = kb*32 + (lane>>4)*16 + h,  N = lane&15
    __shared__ __bf16 ldsB[2 * 32 * 16];

    const int ct   = blockIdx.x;          // column tile: 0..127 (d = ct*16 .. +15)
    const int rblk = blockIdx.y;          // row block:   0..127 (8 row tiles)
    const int tid  = threadIdx.x;
    const int lane = tid & 31;
    const int wave = tid >> 5;

    // ---- Stage B tile (64 x 16 of proj_w -> bf16) into LDS in operand order ----
    // proj_w is small + heavily re-read across blocks: default (RT) caching keeps
    // it hot in L2.
    #pragma unroll
    for (int i = 0; i < 4; ++i) {
        const int e  = tid * 4 + i;       // 0..1023
        const int kb = e >> 9;            // 0..1
        const int l  = (e >> 4) & 31;     // lane slot
        const int h  = e & 15;            // half index within lane
        const int k  = kb * 32 + ((l >> 4) << 4) + h;   // 0..63
        const int n  = l & 15;
        ldsB[e] = (__bf16)proj_w[k * DMODEL + ct * 16 + n];
    }
    __syncthreads();

    // ---- Per-wave row tile ----
    const int rt      = rblk * 8 + wave;  // 0..1023
    const int rowBase = rt * 16;
    const int mrow    = lane & 15;        // A-matrix M index held by this lane
    const int half    = lane >> 4;        // lane half selects K sub-range

    const int    t   = token_ids[rowBase + mrow];
    const float* emb = deep_table + (long)t * KDIM;   // gathered 64-float row

    // A operand layout (16-bit A 16x32): lane half h selects
    //   halfs 0..15 -> K = kb*32 + half*16 + (0..15)
    v16bf a0, a1;
    #pragma unroll
    for (int kb = 0; kb < 2; ++kb) {
        const float* base = emb + kb * 32 + half * 16;
        const float4 c0 = *(const float4*)(base + 0);
        const float4 c1 = *(const float4*)(base + 4);
        const float4 c2 = *(const float4*)(base + 8);
        const float4 c3 = *(const float4*)(base + 12);
        float af[16] = { c0.x, c0.y, c0.z, c0.w,  c1.x, c1.y, c1.z, c1.w,
                         c2.x, c2.y, c2.z, c2.w,  c3.x, c3.y, c3.z, c3.w };
        if (kb == 0) {
            #pragma unroll
            for (int h = 0; h < 16; ++h) a0[h] = (__bf16)af[h];
        } else {
            #pragma unroll
            for (int h = 0; h < 16; ++h) a1[h] = (__bf16)af[h];
        }
    }

    const v16bf b0 = *(const v16bf*)&ldsB[0 * 512 + lane * 16];
    const v16bf b1 = *(const v16bf*)&ldsB[1 * 512 + lane * 16];

    v8f acc = {};
    acc = __builtin_amdgcn_wmma_f32_16x16x32_bf16(
              false, a0, false, b0, (short)0, acc, false, false);
    acc = __builtin_amdgcn_wmma_f32_16x16x32_bf16(
              false, a1, false, b1, (short)0, acc, false, false);

    // ---- Fused epilogue: out = x * (1 + sigmoid(acc)) ----
    // C/D layout: VGPR r -> M = half*8 + r, N = lane&15
    // x/out are streaming touch-once traffic: use non-temporal (TH=NT)
    // accesses so the deep_table + proj_w working set stays resident in L2.
    // sigmoid via hardware v_rcp_f32 (single TRANS op) instead of IEEE divide.
    const int  dcol  = ct * 16 + (lane & 15);
    const int  mbase = half * 8;
    #pragma unroll
    for (int r = 0; r < 8; ++r) {
        const long idx  = (long)(rowBase + mbase + r) * DMODEL + dcol;
        const float gate = __builtin_amdgcn_rcpf(1.0f + __expf(-acc[r]));
        const float xv   = __builtin_nontemporal_load(&x[idx]);
        __builtin_nontemporal_store(xv * (1.0f + gate), &out[idx]);
    }
}

extern "C" void kernel_launch(void* const* d_in, const int* in_sizes, int n_in,
                              void* d_out, int out_size, void* d_ws, size_t ws_size,
                              hipStream_t stream) {
    (void)in_sizes; (void)n_in; (void)d_ws; (void)ws_size; (void)out_size;
    const int*   token_ids  = (const int*)d_in[0];     // [4, 4096] int32
    const float* x          = (const float*)d_in[1];   // [4, 4096, 2048] f32
    const float* deep_table = (const float*)d_in[2];   // [50257, 64] f32
    const float* proj_w     = (const float*)d_in[3];   // [64, 2048] f32
    float*       out        = (float*)d_out;           // [4, 4096, 2048] f32

    dim3 grid(DMODEL / 16, NTOK / (16 * 8));  // (128, 128)
    dim3 block(256);
    deep_embed_wmma_kernel<<<grid, block, 0, stream>>>(
        token_ids, x, deep_table, proj_w, out);
}
